// LSKT_7017976562036
// MI455X (gfx1250) — compile-verified, matrix operand order checked by reading, not run on hardware
//
#include <hip/hip_runtime.h>
#include <math.h>

typedef _Float16 h16;
typedef __attribute__((ext_vector_type(16))) _Float16 v16h;
typedef __attribute__((ext_vector_type(8)))  float    v8f;

#define WMMA_F16(a, b, c) \
  __builtin_amdgcn_wmma_f32_16x16x32_f16(false, (a), false, (b), (short)0, (c), false, false)

// Load a 16x32 (A) / 32x16 (B, fed column-major) f16 fragment from a row-major
// matrix with row stride `ld` (halves). Per ISA 16-bit A layout:
// lane = (half)*16 + r holds row r, K = kb+{0..7} and kb+{16..23}, kb = 8*half.
static __device__ __forceinline__ v16h load_frag16(const h16* base, int ld) {
  const int lane = threadIdx.x & 31;
  const int r  = lane & 15;
  const int kb = (lane >> 4) << 3;
  const h16* p = base + (size_t)r * (size_t)ld + kb;
  union { v16h v; uint4 u[2]; } f;
  f.u[0] = *(const uint4*)(p);
  f.u[1] = *(const uint4*)(p + 16);
  return f.v;
}

// ---------------- converts ----------------
__global__ void cvt_h16(const float* __restrict__ in, h16* __restrict__ out, int n) {
  int i = blockIdx.x * blockDim.x + threadIdx.x;
  if (i < n) out[i] = (h16)in[i];
}

// in[k][n] (512x512 row-major) -> out[n][k] f16
__global__ void cvtT_h16(const float* __restrict__ in, h16* __restrict__ out) {
  const int n = blockIdx.x;
  for (int k = threadIdx.x; k < 512; k += blockDim.x)
    out[(size_t)n * 512 + k] = (h16)in[(size_t)k * 512 + n];
}

// ---------------- GEMM: C[M,512] = A[M,512] * Wt^T + bias ----------------
// One wave computes a 16x64 strip (4 accumulators): A-frag reused 4x per k-step,
// 4 independent WMMAs per iteration. K = N = 512 fixed.
// mode 0: f16 row-major out. mode 1: f16 out transposed per batch [B][512][1024]
// (for V so attention B-frags are contiguous). mode 2: f32 out + residual.
__global__ void __launch_bounds__(128) proj_gemm(
    const h16* __restrict__ A, const h16* __restrict__ Wt,
    const float* __restrict__ bias, void* __restrict__ out,
    const float* __restrict__ resid, int M, int mode) {
  const int wv   = threadIdx.x >> 5;
  const int tile = blockIdx.x * (blockDim.x >> 5) + wv;
  const int tm = tile >> 3;   // 8 strip-tiles along N=512 (64 cols each)
  const int g  = tile & 7;
  if (tm * 16 >= M) return;   // uniform per wave
  const int lane = threadIdx.x & 31;
  const int n0 = g * 64;
  const int c  = lane & 15;

  v8f acc0, acc1, acc2, acc3;
  {
    const float b0 = bias[n0 + c], b1 = bias[n0 + 16 + c];
    const float b2 = bias[n0 + 32 + c], b3 = bias[n0 + 48 + c];
    acc0 = (v8f){b0, b0, b0, b0, b0, b0, b0, b0};
    acc1 = (v8f){b1, b1, b1, b1, b1, b1, b1, b1};
    acc2 = (v8f){b2, b2, b2, b2, b2, b2, b2, b2};
    acc3 = (v8f){b3, b3, b3, b3, b3, b3, b3, b3};
  }
  const h16* a_base = A  + (size_t)tm * 16 * 512;
  const h16* w_base = Wt + (size_t)n0 * 512;
  for (int k = 0; k < 512; k += 32) {
    __builtin_prefetch(a_base + k + 128, 0, 1);
    const v16h af = load_frag16(a_base + k, 512);
    const v16h w0 = load_frag16(w_base + k,             512);
    const v16h w1 = load_frag16(w_base + k + 16 * 512,  512);
    const v16h w2 = load_frag16(w_base + k + 32 * 512,  512);
    const v16h w3 = load_frag16(w_base + k + 48 * 512,  512);
    acc0 = WMMA_F16(af, w0, acc0);
    acc1 = WMMA_F16(af, w1, acc1);
    acc2 = WMMA_F16(af, w2, acc2);
    acc3 = WMMA_F16(af, w3, acc3);
  }
  const int row0 = tm * 16 + ((lane >> 4) << 3);  // this lane's 8 rows
  if (mode == 0) {
    h16* o = (h16*)out;
#pragma unroll
    for (int i = 0; i < 8; ++i) {
      h16* r = o + (size_t)(row0 + i) * 512 + n0 + c;
      r[0]  = (h16)acc0[i];
      r[16] = (h16)acc1[i];
      r[32] = (h16)acc2[i];
      r[48] = (h16)acc3[i];
    }
  } else if (mode == 1) {
    const int bIdx = row0 >> 10;     // 16 | 1024 so tiles never cross batches
    const int s0   = row0 & 1023;
    union { uint4 u; h16 h[8]; } pk0, pk1, pk2, pk3;
#pragma unroll
    for (int i = 0; i < 8; ++i) {
      pk0.h[i] = (h16)acc0[i]; pk1.h[i] = (h16)acc1[i];
      pk2.h[i] = (h16)acc2[i]; pk3.h[i] = (h16)acc3[i];
    }
    h16* o = (h16*)out + ((size_t)bIdx * 512 + n0 + c) * 1024 + s0;
    *(uint4*)(o)               = pk0.u;   // 8 contiguous s-values, 16B aligned
    *(uint4*)(o + 16 * 1024)   = pk1.u;
    *(uint4*)(o + 32 * 1024)   = pk2.u;
    *(uint4*)(o + 48 * 1024)   = pk3.u;
  } else {
    float* o = (float*)out;
#pragma unroll
    for (int i = 0; i < 8; ++i) {
      const size_t idx = (size_t)(row0 + i) * 512 + n0 + c;
      o[idx]      = acc0[i] + resid[idx];
      o[idx + 16] = acc1[i] + resid[idx + 16];
      o[idx + 32] = acc2[i] + resid[idx + 32];
      o[idx + 48] = acc3[i] + resid[idx + 48];
    }
  }
}

// ---------------- flash attention, one wave per (b, h, 16-query tile) --------
__global__ void __launch_bounds__(32) attn_kernel(
    const h16* __restrict__ Qp, const h16* __restrict__ Kp, const h16* __restrict__ Vt,
    const h16* __restrict__ Qr, const h16* __restrict__ Kr,
    const int* __restrict__ smask, const float* __restrict__ gammas,
    h16* __restrict__ outA) {
  const int bid = blockIdx.x;
  const int qt = bid & 63;
  const int h  = (bid >> 6) & 7;
  const int b  = bid >> 9;
  const int lane = threadIdx.x;
  const int lr = lane & 15;
  const int hi = lane >> 4;
  const int q0 = qt << 4;

  // exp(-softplus(g)) == sigmoid(-g); clip lower bound
  const float g = gammas[h];
  float te = 1.0f / (1.0f + __expf(g));
  te = fminf(fmaxf(te, 1e-5f), 1e5f);
  const float sscale = te * 0.0441941738241592f;  // * 1/sqrt(512)
  const float qscale = 0.125f;                    // 1/sqrt(64)

  __shared__ alignas(16) h16 Plds[16 * 32];

  const size_t hd = (size_t)h * 64;
  const h16* qpb = Qp + ((size_t)(b * 1024 + q0) * 512 + hd);
  const h16* qrb = Qr + ((size_t)(b * 1024 + q0) * 512 + hd);
  const v16h aq0 = load_frag16(qpb, 512);
  const v16h aq1 = load_frag16(qpb + 32, 512);
  const v16h ar0 = load_frag16(qrb, 512);
  const v16h ar1 = load_frag16(qrb + 32, 512);

  v8f o0 = {}, o1 = {}, o2 = {}, o3 = {};
  float m8[8], l8[8];
#pragma unroll
  for (int i = 0; i < 8; ++i) { m8[i] = -3.0e38f; l8[i] = 0.0f; }

  const int nb = (q0 + 46) >> 5;  // key blocks of 32 covering keys < q0+16
  for (int kb0 = 0; kb0 < nb * 32; kb0 += 32) {
    float sc[2][8], pr[2][8];
#pragma unroll
    for (int t = 0; t < 2; ++t) {
      const int c0 = kb0 + t * 16;
      const h16* kpb = Kp + ((size_t)(b * 1024 + c0) * 512 + hd);
      const h16* krb = Kr + ((size_t)(b * 1024 + c0) * 512 + hd);
      v8f s = {};
      s = WMMA_F16(aq0, load_frag16(kpb, 512), s);
      s = WMMA_F16(aq1, load_frag16(kpb + 32, 512), s);
      v8f sr = {};
      sr = WMMA_F16(ar0, load_frag16(krb, 512), sr);
      sr = WMMA_F16(ar1, load_frag16(krb + 32, 512), sr);
      const int kc = c0 + lr;
#pragma unroll
      for (int i = 0; i < 8; ++i) {
        const int qr = q0 + i + hi * 8;                 // D-tile row
        const bool causal = kc < qr;                    // strict tril(-1)
        const int mv = smask[((size_t)b * 1024 + qr) * 1024 + kc];
        const float ss = (causal && mv) ? sr[i] * sscale : 0.0f;
        sc[t][i] = causal ? (s[i] * qscale + ss) : -3.0e38f;
      }
    }
    // online softmax (row = 16 lanes within a half-wave + element index)
    float fac[8];
#pragma unroll
    for (int i = 0; i < 8; ++i) {
      float rm = fmaxf(sc[0][i], sc[1][i]);
      rm = fmaxf(rm, __shfl_xor(rm, 1, 32));
      rm = fmaxf(rm, __shfl_xor(rm, 2, 32));
      rm = fmaxf(rm, __shfl_xor(rm, 4, 32));
      rm = fmaxf(rm, __shfl_xor(rm, 8, 32));
      const float mnew = fmaxf(m8[i], rm);
      fac[i] = __expf(m8[i] - mnew);
      pr[0][i] = (sc[0][i] > -1.0e37f) ? __expf(sc[0][i] - mnew) : 0.0f;
      pr[1][i] = (sc[1][i] > -1.0e37f) ? __expf(sc[1][i] - mnew) : 0.0f;
      float rs = pr[0][i] + pr[1][i];
      rs += __shfl_xor(rs, 1, 32);
      rs += __shfl_xor(rs, 2, 32);
      rs += __shfl_xor(rs, 4, 32);
      rs += __shfl_xor(rs, 8, 32);
      l8[i] = l8[i] * fac[i] + rs;
      m8[i] = mnew;
    }
#pragma unroll
    for (int i = 0; i < 8; ++i) {
      o0[i] *= fac[i]; o1[i] *= fac[i]; o2[i] *= fac[i]; o3[i] *= fac[i];
    }
    // transpose P (D-layout) -> A-fragment via LDS
#pragma unroll
    for (int t = 0; t < 2; ++t)
#pragma unroll
      for (int i = 0; i < 8; ++i)
        Plds[(i + hi * 8) * 32 + t * 16 + lr] = (h16)pr[t][i];
    __syncthreads();
    const v16h ap = load_frag16(Plds, 32);
    __syncthreads();
    // P(16x32) @ V(32 keys x 64 dk), V stored transposed [d][s]
    const h16* vb = Vt + ((size_t)(b * 512) + hd) * 1024 + kb0;
    o0 = WMMA_F16(ap, load_frag16(vb,             1024), o0);
    o1 = WMMA_F16(ap, load_frag16(vb + 16 * 1024, 1024), o1);
    o2 = WMMA_F16(ap, load_frag16(vb + 32 * 1024, 1024), o2);
    o3 = WMMA_F16(ap, load_frag16(vb + 48 * 1024, 1024), o3);
  }

  float inv8[8];
#pragma unroll
  for (int i = 0; i < 8; ++i) inv8[i] = (l8[i] > 0.0f) ? (1.0f / l8[i]) : 0.0f;
  h16* ob = outA + ((size_t)(b * 1024 + q0) * 512 + hd);
#pragma unroll
  for (int i = 0; i < 8; ++i) {
    const int r = i + hi * 8;
    ob[(size_t)r * 512 +  0 + lr] = (h16)(o0[i] * inv8[i]);
    ob[(size_t)r * 512 + 16 + lr] = (h16)(o1[i] * inv8[i]);
    ob[(size_t)r * 512 + 32 + lr] = (h16)(o2[i] * inv8[i]);
    ob[(size_t)r * 512 + 48 + lr] = (h16)(o3[i] * inv8[i]);
  }
}

// ---------------- LayerNorm over 512, one 256-thread block per row -----------
__global__ void __launch_bounds__(256) ln_kernel(
    const float* __restrict__ X, const float* __restrict__ gw,
    const float* __restrict__ bw, float* __restrict__ out) {
  const int row = blockIdx.x;
  const float* x = X + (size_t)row * 512;
  const int t = threadIdx.x;
  const float a = x[t], c = x[t + 256];
  __shared__ float red[8];
  float s = a + c;
  for (int m = 1; m < 32; m <<= 1) s += __shfl_xor(s, m, 32);
  if ((t & 31) == 0) red[t >> 5] = s;
  __syncthreads();
  float tot = 0.0f;
#pragma unroll
  for (int i = 0; i < 8; ++i) tot += red[i];
  const float mu = tot * (1.0f / 512.0f);
  __syncthreads();
  const float da = a - mu, dc = c - mu;
  float v = da * da + dc * dc;
  for (int m = 1; m < 32; m <<= 1) v += __shfl_xor(v, m, 32);
  if ((t & 31) == 0) red[t >> 5] = v;
  __syncthreads();
  float vt = 0.0f;
#pragma unroll
  for (int i = 0; i < 8; ++i) vt += red[i];
  const float rstd = rsqrtf(vt * (1.0f / 512.0f) + 1e-5f);
  out[(size_t)row * 512 + t]       = da * rstd * gw[t]       + bw[t];
  out[(size_t)row * 512 + t + 256] = dc * rstd * gw[t + 256] + bw[t + 256];
}

extern "C" void kernel_launch(void* const* d_in, const int* in_sizes, int n_in,
                              void* d_out, int out_size, void* d_ws, size_t ws_size,
                              hipStream_t stream) {
  (void)in_sizes; (void)n_in; (void)out_size; (void)ws_size;
  const float* q_in  = (const float*)d_in[0];
  const float* k_in  = (const float*)d_in[1];
  const float* v_in  = (const float*)d_in[2];
  const int*   smask = (const int*)  d_in[3];
  const float* Wq    = (const float*)d_in[4];
  const float* bq    = (const float*)d_in[5];
  const float* Wv    = (const float*)d_in[6];
  const float* bv    = (const float*)d_in[7];
  const float* Wo    = (const float*)d_in[8];
  const float* bo    = (const float*)d_in[9];
  const float* gm    = (const float*)d_in[10];
  const float* lng   = (const float*)d_in[11];
  const float* lnb   = (const float*)d_in[12];
  float* out = (float*)d_out;

  const size_t E = (size_t)8 * 1024 * 512;  // elements per [B,S,D] tensor
  char* ws = (char*)d_ws;
  h16*  qh  = (h16*)(ws);
  h16*  kh  = (h16*)(ws + 2 * E);
  h16*  vh  = (h16*)(ws + 4 * E);
  h16*  Qp  = (h16*)(ws + 6 * E);
  h16*  Kp  = (h16*)(ws + 8 * E);
  h16*  Vt  = (h16*)(ws + 10 * E);
  h16*  Wqt = (h16*)(ws + 12 * E);
  h16*  Wvt = (h16*)(ws + 12 * E + 524288);
  h16*  Wot = (h16*)(ws + 12 * E + 2 * 524288);
  h16*  Ao  = (h16*)(ws + 12 * E + 3 * 524288);
  float* Xr = (float*)(ws + 12 * E + 3 * 524288 + 2 * E);

  const int n = (int)E;
  cvt_h16<<<dim3((n + 255) / 256), dim3(256), 0, stream>>>(q_in, qh, n);
  cvt_h16<<<dim3((n + 255) / 256), dim3(256), 0, stream>>>(k_in, kh, n);
  cvt_h16<<<dim3((n + 255) / 256), dim3(256), 0, stream>>>(v_in, vh, n);
  cvtT_h16<<<dim3(512), dim3(256), 0, stream>>>(Wq, Wqt);
  cvtT_h16<<<dim3(512), dim3(256), 0, stream>>>(Wv, Wvt);
  cvtT_h16<<<dim3(512), dim3(256), 0, stream>>>(Wo, Wot);

  const int M = 8192;  // B*S rows; (M/16)*(512/64) = 4096 waves, 4 waves/block
  proj_gemm<<<dim3(1024), dim3(128), 0, stream>>>(qh, Wqt, bq, (void*)Qp, nullptr, M, 0);
  proj_gemm<<<dim3(1024), dim3(128), 0, stream>>>(kh, Wqt, bq, (void*)Kp, nullptr, M, 0);
  proj_gemm<<<dim3(1024), dim3(128), 0, stream>>>(vh, Wvt, bv, (void*)Vt, nullptr, M, 1);

  attn_kernel<<<dim3(4096), dim3(32), 0, stream>>>(Qp, Kp, Vt, qh, kh, smask, gm, Ao);

  proj_gemm<<<dim3(1024), dim3(128), 0, stream>>>(Ao, Wot, bo, (void*)Xr, q_in, M, 2);
  ln_kernel<<<dim3(8192), dim3(256), 0, stream>>>(Xr, lng, lnb, out);
}
